// TF_60370060312584
// MI455X (gfx1250) — matrix operand order repaired and checked by reading, not compiled, not run on hardware
//
#include <hip/hip_runtime.h>

typedef _Float16 f16;
typedef __attribute__((ext_vector_type(16))) _Float16 v16h;
typedef __attribute__((ext_vector_type(8)))  float    v8f;
typedef unsigned int u32;
typedef unsigned long long u64;

#define NBROWS 8192   // N*B
#define DMODEL 512
#define LSEQ   1024
#define NBATCH 8
#define NHEAD  8
#define HDIM   64
#define DFF    2048

// LDS tile: 128 rows x 16 f16-pairs, row stride 20 uints (80B: 16B-aligned for
// async b128 writes; 20n mod 64 has period 16 -> 16 consecutive rows hit 16
// distinct banks). One buffer = A(2560 u32) + B(2560 u32); double buffered.
#define ROWSTRIDE 20
#define TILEU32   (128 * ROWSTRIDE)      // 2560
#define BUFU32    (2 * TILEU32)          // 5120

union FragU { u32 u[8]; v16h v; };

__device__ __forceinline__ float lrelu(float v) { return v >= 0.f ? v : 0.2f * v; }

// CDNA5 async global->LDS copy, 16B per lane, tracked by ASYNCcnt.
__device__ __forceinline__ void async_ld16(u32 lds_byte_addr, const void* g)
{
  asm volatile("global_load_async_to_lds_b128 %0, %1, off"
               :: "v"(lds_byte_addr), "v"((u64)(uintptr_t)g) : "memory");
}
__device__ __forceinline__ void wait_async0()
{
#if __has_builtin(__builtin_amdgcn_s_wait_asynccnt)
  __builtin_amdgcn_s_wait_asynccnt(0);
#else
  asm volatile("s_wait_asynccnt 0" ::: "memory");
#endif
}

// ---------------------------------------------------------------------------
// Generic WMMA GEMM:  Out[m,n] = act( A[m,:] . Wt[n,:] + bias[n] )
// A: f16 [M x K] row-major (lda), Wt: f16 [N x K] row-major (ldw).
// Tile 128x128, BK=32, double-buffered async-to-LDS staging.
// 8 waves: 4x2 wave grid, each wave 32x64 (2x4 WMMA accumulators).
// OUT_MODE: 0 = f32 row-major (ldc,colofs), 1 = f16 row-major (ldc,colofs),
//           2 = f32 QKV layout [b][h][seq][d]  (b=m%8, seq=m/8, h=n/64, d=n%64)
// ---------------------------------------------------------------------------
template <int OUT_MODE>
__global__ __launch_bounds__(256) void k_gemm(
    const f16* __restrict__ A, int lda,
    const f16* __restrict__ Wt, int ldw,
    const float* __restrict__ bias,
    void* __restrict__ Out, int ldc, int colofs,
    int K, int act)
{
  __shared__ u32 lds[2 * BUFU32];            // 40 KB

  const int tid  = threadIdx.x;
  const int row0 = blockIdx.y * 128;
  const int col0 = blockIdx.x * 128;
  const int wave = tid >> 5, lane = tid & 31;
  const int wm = wave >> 1, wn = wave & 1;   // 4x2 wave grid
  const int hf = lane >> 4, l16 = lane & 15;
  const u32 ldsbase = (u32)(uintptr_t)&lds[0];   // flat addr low32 == LDS offset

  // Issue one 128x32 tile of A and of W into LDS buffer `buf` (async, 4 ops/thread).
  auto issue_tile = [&](int k0, int buf) {
#pragma unroll
    for (int it = 0; it < 2; ++it) {
      int p = tid + it * 256;                // 512 x 16B chunks per matrix
      int m = p >> 2, kc = p & 3;
      u32 la = ldsbase + (u32)((buf * BUFU32 + m * ROWSTRIDE + kc * 4) * 4);
      async_ld16(la,                 A  + (size_t)(row0 + m) * lda + k0 + kc * 8);
      async_ld16(la + TILEU32 * 4,   Wt + (size_t)(col0 + m) * ldw + k0 + kc * 8);
    }
  };

  v8f acc[2][4];
#pragma unroll
  for (int a2 = 0; a2 < 2; ++a2)
#pragma unroll
    for (int b2 = 0; b2 < 4; ++b2)
      acc[a2][b2] = (v8f){0.f,0.f,0.f,0.f,0.f,0.f,0.f,0.f};

  const int T = K >> 5;
  issue_tile(0, 0);

  for (int t = 0; t < T; ++t) {
    wait_async0();                 // tile t resident (async loads retire in order)
    __syncthreads();               // ...in every wave
    if (t + 1 < T) issue_tile((t + 1) << 5, (t + 1) & 1);  // overlap with compute

    const u32* Ab = lds + (t & 1) * BUFU32;
    const u32* Bb = Ab + TILEU32;

    FragU af[2], bfr[4];
#pragma unroll
    for (int ai = 0; ai < 2; ++ai) {         // A 16x32: lane<16 K={0..7,16..23}
      int m = wm * 32 + ai * 16 + l16;
#pragma unroll
      for (int i = 0; i < 8; ++i)
        af[ai].u[i] = Ab[m * ROWSTRIDE + ((i >> 2) * 8 + hf * 4 + (i & 3))];
    }
#pragma unroll
    for (int bi = 0; bi < 4; ++bi) {         // B 32x16: lane<16 K={0..7,16..23}
      int n = wn * 64 + bi * 16 + l16;
#pragma unroll
      for (int i = 0; i < 8; ++i)
        bfr[bi].u[i] = Bb[n * ROWSTRIDE + ((i >> 2) * 8 + hf * 4 + (i & 3))];
    }
#pragma unroll
    for (int ai = 0; ai < 2; ++ai)
#pragma unroll
      for (int bi = 0; bi < 4; ++bi)
        acc[ai][bi] = __builtin_amdgcn_wmma_f32_16x16x32_f16(
            false, af[ai].v, false, bfr[bi].v, (short)0, acc[ai][bi], false, false);
    // Next iteration's wait+barrier protects LDS reuse (frag reads completed
    // before each wave's WMMAs via compiler dscnt waits).
  }

  // Hoist bias (one value per (bi, lane) pair, independent of ai/j).
  float bv[4] = {0.f, 0.f, 0.f, 0.f};
  if (bias) {
#pragma unroll
    for (int bi = 0; bi < 4; ++bi) bv[bi] = bias[col0 + wn * 64 + bi * 16 + l16];
  }

#pragma unroll
  for (int ai = 0; ai < 2; ++ai)
#pragma unroll
    for (int bi = 0; bi < 4; ++bi)
#pragma unroll
      for (int j = 0; j < 8; ++j) {
        int gm = row0 + wm * 32 + ai * 16 + hf * 8 + j;  // C/D: M = j + half*8
        int gn = col0 + wn * 64 + bi * 16 + l16;         //      N = lane&15
        float v = acc[ai][bi][j] + bv[bi];
        if (act == 1) v = fmaxf(v, 0.f);
        if constexpr (OUT_MODE == 0) {
          ((float*)Out)[(size_t)gm * ldc + colofs + gn] = v;
        } else if constexpr (OUT_MODE == 1) {
          ((f16*)Out)[(size_t)gm * ldc + colofs + gn] = (f16)v;
        } else {
          int b = gm & 7, seq = gm >> 3, hh = gn >> 6, d = gn & 63;
          ((float*)Out)[(size_t)((b * 8 + hh) * 1024 + seq) * 64 + d] = v;
        }
      }
}

// ---------------------------------------------------------------------------
// Softmax attention: one block per (query, batch-head). qp/kp/vp [bh][seq][64].
// out row-major [seq*8 + b][h*64 + d].
// ---------------------------------------------------------------------------
__global__ __launch_bounds__(256) void k_attn(
    const float* __restrict__ qp, const float* __restrict__ kp,
    const float* __restrict__ vp, float* __restrict__ out)
{
  __shared__ float qs[64];
  __shared__ float sc[1024];
  __shared__ float red[256];
  __shared__ float op[4][64];
  const int tid = threadIdx.x;
  const int qi = blockIdx.x, bh = blockIdx.y;
  const size_t base = (size_t)bh * (LSEQ * HDIM);

  if (tid < 64) qs[tid] = qp[base + (size_t)qi * 64 + tid];
  __syncthreads();

  float s[4]; float lmax = -3.4e38f;
#pragma unroll
  for (int t = 0; t < 4; ++t) {
    int ki = tid + t * 256;
    const float4* kr4 = (const float4*)(kp + base + (size_t)ki * 64);
    float a = 0.f;
#pragma unroll
    for (int d4 = 0; d4 < 16; ++d4) {
      float4 kv = kr4[d4];
      a += qs[d4 * 4 + 0] * kv.x + qs[d4 * 4 + 1] * kv.y
         + qs[d4 * 4 + 2] * kv.z + qs[d4 * 4 + 3] * kv.w;
    }
    a *= 0.125f;                                  // 1/sqrt(64)
    s[t] = a; lmax = fmaxf(lmax, a);
  }
  red[tid] = lmax; __syncthreads();
  for (int st = 128; st; st >>= 1) { if (tid < st) red[tid] = fmaxf(red[tid], red[tid + st]); __syncthreads(); }
  float gmax = red[0]; __syncthreads();

  float lsum = 0.f;
#pragma unroll
  for (int t = 0; t < 4; ++t) { float e = __expf(s[t] - gmax); sc[tid + t * 256] = e; lsum += e; }
  red[tid] = lsum; __syncthreads();
  for (int st = 128; st; st >>= 1) { if (tid < st) red[tid] += red[tid + st]; __syncthreads(); }
  float inv = 1.f / red[0];

  int d = tid & 63, chunk = tid >> 6;
  float a = 0.f;
  for (int k2 = chunk * 256; k2 < chunk * 256 + 256; ++k2)
    a += sc[k2] * vp[base + (size_t)k2 * 64 + d];
  op[chunk][d] = a; __syncthreads();
  if (tid < 64) {
    float o = (op[0][tid] + op[1][tid] + op[2][tid] + op[3][tid]) * inv;
    out[(size_t)(qi * 8 + (bh >> 3)) * 512 + (bh & 7) * 64 + tid] = o;
  }
}

// t3 = max over 3x3 clipped neighbors of lrelu(Ak[nb,b,c] + Aq[n,b,c]); write f16
// into concat buffer at column 512+c (ldc=1024).
__global__ __launch_bounds__(256) void k_graph_max(
    const float* __restrict__ Aq, const float* __restrict__ Ak, f16* __restrict__ cat)
{
  int gid = blockIdx.x * 256 + threadIdx.x;     // over NBROWS*512
  int r = gid >> 9, c = gid & 511;
  int n = r >> 3, b = r & 7;
  int i = n >> 5, j = n & 31;
  float aq = Aq[gid];
  float mx = -3.4e38f;
#pragma unroll
  for (int di = -1; di <= 1; ++di) {
    int ii = min(max(i + di, 0), 31);
#pragma unroll
    for (int dj = -1; dj <= 1; ++dj) {
      int jj = min(max(j + dj, 0), 31);
      int nb = ii * 32 + jj;
      float v = Ak[(size_t)(nb * 8 + b) * 512 + c] + aq;
      mx = fmaxf(mx, lrelu(v));
    }
  }
  cat[(size_t)r * 1024 + 512 + c] = (f16)mx;
}

// out = LayerNorm(x + t2) * g + b   (one block per row, D=512, in-place safe)
__global__ __launch_bounds__(256) void k_ln(
    const float* __restrict__ x, const float* __restrict__ t2,
    const float* __restrict__ g, const float* __restrict__ bta, float* __restrict__ o)
{
  __shared__ float red[256];
  int r = blockIdx.x, tid = threadIdx.x;
  size_t base = (size_t)r * 512;
  float v0 = x[base + tid]       + t2[base + tid];
  float v1 = x[base + tid + 256] + t2[base + tid + 256];
  red[tid] = v0 + v1; __syncthreads();
  for (int s = 128; s; s >>= 1) { if (tid < s) red[tid] += red[tid + s]; __syncthreads(); }
  float mean = red[0] * (1.f / 512.f); __syncthreads();
  float d0 = v0 - mean, d1 = v1 - mean;
  red[tid] = d0 * d0 + d1 * d1; __syncthreads();
  for (int s = 128; s; s >>= 1) { if (tid < s) red[tid] += red[tid + s]; __syncthreads(); }
  float rstd = rsqrtf(red[0] * (1.f / 512.f) + 1e-5f);
  o[base + tid]       = d0 * rstd * g[tid]       + bta[tid];
  o[base + tid + 256] = d1 * rstd * g[tid + 256] + bta[tid + 256];
}

// o[i] = f16(a[i] + b[i])  (b may be null -> plain convert)
__global__ void k_add_cvt(const float* __restrict__ a, const float* __restrict__ b,
                          f16* __restrict__ o, int n)
{
  int i = blockIdx.x * 256 + threadIdx.x;
  if (i < n) o[i] = (f16)(b ? a[i] + b[i] : a[i]);
}

// W1 = proj_w[:, :512]  (proj_w is [512 x 1024] row-major)
__global__ void k_cvt_w1(const float* __restrict__ w, f16* __restrict__ o)
{
  int i = blockIdx.x * 256 + threadIdx.x;       // 512*512
  int r = i >> 9, c = i & 511;
  o[i] = (f16)w[r * 1024 + c];
}
// Wdiff = proj_w[:, 512:] - proj_w[:, :512]
__global__ void k_cvt_wdiff(const float* __restrict__ w, f16* __restrict__ o)
{
  int i = blockIdx.x * 256 + threadIdx.x;
  int r = i >> 9, c = i & 511;
  o[i] = (f16)(w[r * 1024 + 512 + c] - w[r * 1024 + c]);
}

extern "C" void kernel_launch(void* const* d_in, const int* in_sizes, int n_in,
                              void* d_out, int out_size, void* d_ws, size_t ws_size,
                              hipStream_t stream)
{
  (void)in_sizes; (void)n_in; (void)out_size; (void)ws_size;
  const float* tgt      = (const float*)d_in[0];
  const float* memory   = (const float*)d_in[1];
  const float* pos      = (const float*)d_in[2];
  const float* sa_in_w  = (const float*)d_in[3];
  const float* sa_in_b  = (const float*)d_in[4];
  const float* sa_out_w = (const float*)d_in[5];
  const float* sa_out_b = (const float*)d_in[6];
  const float* ca_in_w  = (const float*)d_in[7];
  const float* ca_in_b  = (const float*)d_in[8];
  const float* ca_out_w = (const float*)d_in[9];
  const float* ca_out_b = (const float*)d_in[10];
  const float* proj_w   = (const float*)d_in[11];
  const float* proj_b   = (const float*)d_in[12];
  const float* proj2_w  = (const float*)d_in[13];
  const float* proj2_b  = (const float*)d_in[14];
  const float* projc_w  = (const float*)d_in[15];
  const float* projc_b  = (const float*)d_in[16];
  const float* proj2c_w = (const float*)d_in[17];
  const float* proj2c_b = (const float*)d_in[18];
  const float* lin1_w   = (const float*)d_in[19];
  const float* lin1_b   = (const float*)d_in[20];
  const float* lin2_w   = (const float*)d_in[21];
  const float* lin2_b   = (const float*)d_in[22];
  const float* n1_g = (const float*)d_in[23];
  const float* n1_b = (const float*)d_in[24];
  const float* n2_g = (const float*)d_in[25];
  const float* n2_b = (const float*)d_in[26];
  const float* n3_g = (const float*)d_in[27];
  const float* n3_b = (const float*)d_in[28];
  float* out = (float*)d_out;

  // ---- workspace bump allocator --------------------------------------------
  char* p = (char*)d_ws;
  auto alloc = [&](size_t bytes) -> char* {
    char* r = p; p += (bytes + 255) & ~(size_t)255; return r;
  };
  const size_t NBD = (size_t)NBROWS * DMODEL;
  // f16 weights
  f16* sa_in_h  = (f16*)alloc(1536 * 512 * 2);
  f16* sa_out_h = (f16*)alloc(512 * 512 * 2);
  f16* ca_in_h  = (f16*)alloc(1536 * 512 * 2);
  f16* ca_out_h = (f16*)alloc(512 * 512 * 2);
  f16* w1_h     = (f16*)alloc(512 * 512 * 2);
  f16* wd_h     = (f16*)alloc(512 * 512 * 2);
  f16* w1c_h    = (f16*)alloc(512 * 512 * 2);
  f16* wdc_h    = (f16*)alloc(512 * 512 * 2);
  f16* p2_h     = (f16*)alloc(512 * 1024 * 2);
  f16* p2c_h    = (f16*)alloc(512 * 1024 * 2);
  f16* l1_h     = (f16*)alloc(2048 * 512 * 2);
  f16* l2_h     = (f16*)alloc(512 * 2048 * 2);
  // f16 activations (x_h..memp_h contiguous: reused as FFN hidden in stage 3)
  f16* x_h    = (f16*)alloc(NBD * 2);
  f16* xp_h   = (f16*)alloc(NBD * 2);
  f16* mem_h  = (f16*)alloc(NBD * 2);
  f16* memp_h = (f16*)alloc(NBD * 2);
  f16* attn_h = (f16*)alloc(NBD * 2);
  f16* cat_h  = (f16*)alloc(NBD * 2 * 2);        // [8192 x 1024]
  f16* h_h    = x_h;                              // [8192 x 2048] alias (stage 3)
  // f32 buffers
  float* qp     = (float*)alloc(NBD * 4);
  float* kp     = (float*)alloc(NBD * 4);
  float* vp     = (float*)alloc(NBD * 4);
  float* attn_f = (float*)alloc(NBD * 4);
  float* t2     = (float*)alloc(NBD * 4);
  float* xf     = (float*)alloc(NBD * 4);
  float* Ak = qp;   // reused after attention is done
  float* Aq = kp;

  auto cvt = [&](const float* a, const float* b, f16* o, int n) {
    k_add_cvt<<<(n + 255) / 256, 256, 0, stream>>>(a, b, o, n);
  };
  const dim3 g512(4, 64), g2048(16, 64);
  const dim3 gattn(1024, 64);
  const int nAct = (int)NBD;

  // ---- weight conversions ---------------------------------------------------
  cvt(sa_in_w,  nullptr, sa_in_h,  1536 * 512);
  cvt(sa_out_w, nullptr, sa_out_h, 512 * 512);
  cvt(ca_in_w,  nullptr, ca_in_h,  1536 * 512);
  cvt(ca_out_w, nullptr, ca_out_h, 512 * 512);
  k_cvt_w1   <<<1024, 256, 0, stream>>>(proj_w,  w1_h);
  k_cvt_wdiff<<<1024, 256, 0, stream>>>(proj_w,  wd_h);
  k_cvt_w1   <<<1024, 256, 0, stream>>>(projc_w, w1c_h);
  k_cvt_wdiff<<<1024, 256, 0, stream>>>(projc_w, wdc_h);
  cvt(proj2_w,  nullptr, p2_h,  512 * 1024);
  cvt(proj2c_w, nullptr, p2c_h, 512 * 1024);
  cvt(lin1_w,   nullptr, l1_h,  2048 * 512);
  cvt(lin2_w,   nullptr, l2_h,  512 * 2048);

  // ================= Stage 1: self-attention + graph =========================
  cvt(tgt, pos,     xp_h, nAct);                 // q,k source = x+pos
  cvt(tgt, nullptr, x_h,  nAct);                 // v / graph source = x
  k_gemm<2><<<g512, 256, 0, stream>>>(xp_h, 512, sa_in_h,              512, sa_in_b,        qp, 0, 0, 512, 0);
  k_gemm<2><<<g512, 256, 0, stream>>>(xp_h, 512, sa_in_h +  512 * 512, 512, sa_in_b + 512,  kp, 0, 0, 512, 0);
  k_gemm<2><<<g512, 256, 0, stream>>>(x_h,  512, sa_in_h + 1024 * 512, 512, sa_in_b + 1024, vp, 0, 0, 512, 0);
  k_attn<<<gattn, 256, 0, stream>>>(qp, kp, vp, attn_f);
  cvt(attn_f, nullptr, attn_h, nAct);
  k_gemm<1><<<g512, 256, 0, stream>>>(attn_h, 512, sa_out_h, 512, sa_out_b, cat_h, 1024, 0, 512, 0); // t2 -> cat[:, :512]
  k_gemm<0><<<g512, 256, 0, stream>>>(x_h, 512, w1_h, 512, nullptr, Ak, 512, 0, 512, 0);   // k @ W1.T
  k_gemm<0><<<g512, 256, 0, stream>>>(x_h, 512, wd_h, 512, proj_b,  Aq, 512, 0, 512, 0);   // q @ (W2-W1).T + b
  k_graph_max<<<(nAct + 255) / 256, 256, 0, stream>>>(Aq, Ak, cat_h);                      // -> cat[:, 512:]
  k_gemm<0><<<g512, 256, 0, stream>>>(cat_h, 1024, p2_h, 1024, proj2_b, t2, 512, 0, 1024, 0);
  k_ln<<<NBROWS, 256, 0, stream>>>(tgt, t2, n1_g, n1_b, xf);

  // ================= Stage 2: cross-attention + graph ========================
  cvt(xf, pos,        xp_h,   nAct);             // q source = x+pos
  cvt(memory, pos,    memp_h, nAct);             // k source = memory+pos
  cvt(memory, nullptr, mem_h, nAct);             // v / graph-k source = memory
  k_gemm<2><<<g512, 256, 0, stream>>>(xp_h,   512, ca_in_h,              512, ca_in_b,        qp, 0, 0, 512, 0);
  k_gemm<2><<<g512, 256, 0, stream>>>(memp_h, 512, ca_in_h +  512 * 512, 512, ca_in_b + 512,  kp, 0, 0, 512, 0);
  k_gemm<2><<<g512, 256, 0, stream>>>(mem_h,  512, ca_in_h + 1024 * 512, 512, ca_in_b + 1024, vp, 0, 0, 512, 0);
  k_attn<<<gattn, 256, 0, stream>>>(qp, kp, vp, attn_f);
  cvt(attn_f, nullptr, attn_h, nAct);
  k_gemm<1><<<g512, 256, 0, stream>>>(attn_h, 512, ca_out_h, 512, ca_out_b, cat_h, 1024, 0, 512, 0);
  cvt(xf, nullptr, x_h, nAct);
  k_gemm<0><<<g512, 256, 0, stream>>>(mem_h, 512, w1c_h, 512, nullptr, Ak, 512, 0, 512, 0); // memory @ W1c.T
  k_gemm<0><<<g512, 256, 0, stream>>>(x_h,   512, wdc_h, 512, projc_b, Aq, 512, 0, 512, 0); // x @ (W2c-W1c).T
  k_graph_max<<<(nAct + 255) / 256, 256, 0, stream>>>(Aq, Ak, cat_h);
  k_gemm<0><<<g512, 256, 0, stream>>>(cat_h, 1024, p2c_h, 1024, proj2c_b, t2, 512, 0, 1024, 0);
  k_ln<<<NBROWS, 256, 0, stream>>>(xf, t2, n2_g, n2_b, xf);

  // ================= Stage 3: FFN ============================================
  cvt(xf, nullptr, attn_h, nAct);                // f16(x2) (attn_h is free now)
  k_gemm<1><<<g2048, 256, 0, stream>>>(attn_h, 512, l1_h, 512, lin1_b, h_h, 2048, 0, 512, 1);  // relu -> f16 hidden
  k_gemm<0><<<g512,  256, 0, stream>>>(h_h,   2048, l2_h, 2048, lin2_b, t2, 512, 0, 2048, 0);
  k_ln<<<NBROWS, 256, 0, stream>>>(xf, t2, n3_g, n3_b, out);
}
